// BiLSTM_9028021256417
// MI455X (gfx1250) — compile-verified, matrix operand order checked by reading, not run on hardware
//
#include <hip/hip_runtime.h>

typedef __attribute__((ext_vector_type(16))) __bf16 v16bf;
typedef __attribute__((ext_vector_type(8)))  float  v8f;
typedef __attribute__((ext_vector_type(8)))  unsigned short u16x8;
typedef __attribute__((ext_vector_type(4)))  unsigned short u16x4;
typedef __attribute__((ext_vector_type(4)))  float  f32x4;

#define BB   512
#define TT   200
#define EE   128
#define HH   128
#define G4   512   // 4*H
#define KTOT 256   // E + H
#define MT   16    // batch rows per workgroup
#define NT   4     // 16-wide N tiles per wave (8 waves cover 512 gate cols)
#define KC   8     // K chunks of 32 (256 total)

__device__ __forceinline__ unsigned short f32_to_bf16(float f) {
  unsigned int x = __float_as_uint(f);
  x += 0x7FFFu + ((x >> 16) & 1u);       // round-to-nearest-even
  return (unsigned short)(x >> 16);
}

union BF16Frag { v16bf bf; u16x8 u8[2]; };

// Pre-swizzle [W;U] (f32) into bf16 WMMA B-fragment layout:
// wfrag[pass][kc][ntile][lane][e], k = kc*32 + (e>>3)*16 + (lane>>4)*8 + (e&7),
// n = ntile*16 + (lane&15).  524288 elements total (4 passes).
__global__ __launch_bounds__(256) void prep_weights(
    const float* __restrict__ W, const float* __restrict__ U,
    unsigned short* __restrict__ wfrag)
{
  int idx   = blockIdx.x * 256 + threadIdx.x;
  int e     = idx & 15;
  int lane  = (idx >> 4) & 31;
  int ntile = (idx >> 9) & 31;
  int kc    = (idx >> 14) & 7;
  int pass  = idx >> 17;                 // layer*2 + dir
  int k = kc * 32 + ((e >> 3) << 4) + ((lane >> 4) << 3) + (e & 7);
  int n = ntile * 16 + (lane & 15);
  float v = (k < EE) ? W[((size_t)pass * EE + k) * G4 + n]
                     : U[((size_t)pass * HH + (k - EE)) * G4 + n];
  wfrag[idx] = f32_to_bf16(v);
}

// Fused fw+bw LSTM layer: blocks 0..31 run the forward chain, 32..63 the
// backward chain (independent recurrences -> 2x WGP utilization, half the
// sequential chain).  z = [x_t, h] @ [W;U] via v_wmma_f32_16x16x32_bf16.
// All B fragments (256 VGPRs) live in the high VGPR file across the T loop.
// Next-step x is prefetched into registers under the WMMA phase.
// out[b,t,j] = h + addIn*in[b,t,j]   (in-place out==in is safe per-step)
__global__ __launch_bounds__(256, 1) void lstm_dual(
    const float* __restrict__ inF, const float* __restrict__ inB,
    const unsigned short* __restrict__ wfF, const unsigned short* __restrict__ wfB,
    const float* __restrict__ biasF, const float* __restrict__ biasB,
    float* __restrict__ outF, float* __restrict__ outB, int addIn)
{
  const int dir = blockIdx.x >> 5;             // 0 = forward, 1 = backward
  const int b0  = (blockIdx.x & 31) * MT;
  const float* __restrict__ in   = dir ? inB   : inF;
  const unsigned short* __restrict__ wfrag = dir ? wfB : wfF;
  const float* __restrict__ bias = dir ? biasB : biasF;
  float* __restrict__ out        = dir ? outB  : outF;

  __shared__ __align__(16) unsigned short xh[MT][KTOT]; // bf16 [x_t | h], 8KB
  __shared__ float zbuf[MT][G4];                        // gate pre-acts, 32KB
  __shared__ float cbuf[MT][HH];                        // cell state, 8KB
  __shared__ float bl[G4];                              // bias, 2KB

  const int tid  = threadIdx.x;
  const int lane = tid & 31;
  const int wave = tid >> 5;        // owns gate columns [wave*64, wave*64+64)
  const int m16  = lane & 15;
  const int hi   = lane >> 4;

  // h = 0, c = 0, bias -> LDS
  for (int i = tid; i < MT * HH; i += 256) {
    cbuf[i >> 7][i & 127] = 0.0f;
    xh[i >> 7][(i & 127) + HH] = 0;
  }
  for (int i = tid; i < G4; i += 256) bl[i] = bias[i];

  // Hoist ALL B fragments (8 kc x 4 nt x 8 VGPR = 256 VGPRs) into registers.
  BF16Frag Breg[KC][NT];
#pragma unroll
  for (int kc = 0; kc < KC; ++kc)
#pragma unroll
    for (int n = 0; n < NT; ++n) {
      const u16x8* bp = (const u16x8*)
          (wfrag + (((size_t)kc * 32 + wave * NT + n) * 32 + lane) * 16);
      Breg[kc][n].u8[0] = bp[0];
      Breg[kc][n].u8[1] = bp[1];
    }

  // Prologue: stage x at the first timestep (vectorized b128 -> 4x bf16)
  {
    const int t0 = dir ? TT - 1 : 0;
#pragma unroll
    for (int it = 0; it < 2; ++it) {
      int i4 = tid + it * 256;                 // 512 float4 units = 16x128
      int m = i4 >> 5, e = (i4 & 31) * 4;
      f32x4 v = *(const f32x4*)&in[((size_t)(b0 + m) * TT + t0) * EE + e];
      u16x4 p;
      p[0] = f32_to_bf16(v[0]); p[1] = f32_to_bf16(v[1]);
      p[2] = f32_to_bf16(v[2]); p[3] = f32_to_bf16(v[3]);
      *(u16x4*)&xh[m][e] = p;
    }
  }
  __syncthreads();

  for (int step = 0; step < TT; ++step) {
    const int t = dir ? (TT - 1 - step) : step;

    // ---- z[0:16, wave*64 : +64] = [x_t,h] @ [W;U] slice ----
    v8f acc[NT] = {};
#pragma unroll
    for (int kc = 0; kc < KC; ++kc) {
      BF16Frag A;                        // ISA 16-bit 16x32 A layout
      int kbase = kc * 32 + hi * 8;
      A.u8[0] = *(const u16x8*)&xh[m16][kbase];
      A.u8[1] = *(const u16x8*)&xh[m16][kbase + 16];
#pragma unroll
      for (int n = 0; n < NT; ++n)
        acc[n] = __builtin_amdgcn_wmma_f32_16x16x32_bf16(
            false, A.bf, false, Breg[kc][n].bf, (short)0, acc[n], false, false);
    }

    // ---- prefetch next timestep's x into registers (hides under WMMA) ----
    const bool hasNext = (step + 1 < TT);
    f32x4 xp[2];
    int pm[2], pe[2];
    if (hasNext) {
      const int tn = dir ? (t - 1) : (t + 1);
#pragma unroll
      for (int it = 0; it < 2; ++it) {
        int i4 = tid + it * 256;
        pm[it] = i4 >> 5; pe[it] = (i4 & 31) * 4;
        xp[it] = *(const f32x4*)&in[((size_t)(b0 + pm[it]) * TT + tn) * EE + pe[it]];
      }
    }

    // ---- spill C/D fragments (VGPR r: M = r + 8*hi, N = lane%16) ----
#pragma unroll
    for (int n = 0; n < NT; ++n) {
      int col = wave * (NT * 16) + n * 16 + m16;
#pragma unroll
      for (int r = 0; r < 8; ++r)
        zbuf[hi * 8 + r][col] = acc[n][r];
    }
    __syncthreads();   // all A-reads of xh done; z visible

    // commit prefetched x for the next step
    if (hasNext) {
#pragma unroll
      for (int it = 0; it < 2; ++it) {
        u16x4 p;
        p[0] = f32_to_bf16(xp[it][0]); p[1] = f32_to_bf16(xp[it][1]);
        p[2] = f32_to_bf16(xp[it][2]); p[3] = f32_to_bf16(xp[it][3]);
        *(u16x4*)&xh[pm[it]][pe[it]] = p;
      }
    }

    // ---- gates + state update + residual epilogue ----
    for (int i = tid; i < MT * HH; i += 256) {
      int m = i >> 7, j = i & 127;
      float zi = zbuf[m][j]       + bl[j];
      float zf = zbuf[m][j + 128] + bl[j + 128];
      float zg = zbuf[m][j + 256] + bl[j + 256];
      float zo = zbuf[m][j + 384] + bl[j + 384];
      float ig = 1.0f / (1.0f + __expf(-zi));
      float fg = 1.0f / (1.0f + __expf(-zf));
      float gg = tanhf(zg);
      float og = 1.0f / (1.0f + __expf(-zo));
      float c  = fg * cbuf[m][j] + ig * gg;
      cbuf[m][j] = c;
      float h  = og * tanhf(c);
      xh[m][HH + j] = f32_to_bf16(h);    // recurrent input for next step
      size_t oidx = ((size_t)(b0 + m) * TT + t) * HH + j;
      float ov = h;
      if (addIn) ov += in[oidx];         // layer-1 residual (read-before-write)
      out[oidx] = ov;
    }
    __syncthreads();
  }
}

// out = 0.5 * (fw1 + bw1), pure bandwidth (~105MB -> ~4.5us at 23.3 TB/s)
__global__ __launch_bounds__(256) void combine_avg(
    const f32x4* __restrict__ a, const f32x4* __restrict__ b,
    f32x4* __restrict__ o, int n4)
{
  int i = blockIdx.x * 256 + threadIdx.x;
  if (i < n4) {
    f32x4 x = a[i], y = b[i], r;
    r[0] = 0.5f * (x[0] + y[0]); r[1] = 0.5f * (x[1] + y[1]);
    r[2] = 0.5f * (x[2] + y[2]); r[3] = 0.5f * (x[3] + y[3]);
    o[i] = r;
  }
}

extern "C" void kernel_launch(void* const* d_in, const int* in_sizes, int n_in,
                              void* d_out, int out_size, void* d_ws, size_t ws_size,
                              hipStream_t stream) {
  const float* x  = (const float*)d_in[0];  // [512,200,128]
  const float* W  = (const float*)d_in[1];  // [2,2,128,512]
  const float* U  = (const float*)d_in[2];  // [2,2,128,512]
  const float* bb = (const float*)d_in[3];  // [2,2,512]
  float* out = (float*)d_out;

  const size_t seq = (size_t)BB * TT * HH;               // 13.1M floats
  float* fw0 = (float*)d_ws;
  float* bw0 = fw0 + seq;
  unsigned short* wfrag = (unsigned short*)(bw0 + seq);  // 4 x 256KB bf16

  prep_weights<<<2048, 256, 0, stream>>>(W, U, wfrag);

  const size_t pW = (size_t)KC * 32 * 32 * 16;           // 131072 ushorts/pass
  // layer 0: fw+bw chains concurrently (64 WGs)
  lstm_dual<<<64, 256, 0, stream>>>(x, x, wfrag, wfrag + pW,
                                    bb, bb + G4, fw0, bw0, 0);
  // layer 1 (+residual), in-place on fw0/bw0 (64 WGs)
  lstm_dual<<<64, 256, 0, stream>>>(fw0, bw0, wfrag + 2 * pW, wfrag + 3 * pW,
                                    bb + 2 * G4, bb + 3 * G4, fw0, bw0, 1);
  // final average
  const int n4 = (int)(seq / 4);
  combine_avg<<<(n4 + 255) / 256, 256, 0, stream>>>(
      (const f32x4*)fw0, (const f32x4*)bw0, (f32x4*)out, n4);
}
